// VeinCapsNet_34686155883106
// MI455X (gfx1250) — matrix operand order repaired and checked by reading, not compiled, not run on hardware
//
#include <hip/hip_runtime.h>
#include <hip/hip_bf16.h>
#include <stdint.h>

// ---------------------------------------------------------------------------
// CapsNet forward for MI455X (gfx1250, wave32, WMMA).
// u_hat (508MB) is never materialized; it is recomputed with
// v_wmma_f32_16x16x32_bf16 inside each routing pass. Routing logits are
// eliminated algebraically (b2 = u_hat·(v0+v1)). W[r] is staged into the
// 320KB WGP LDS with global_load_async_to_lds_b128. 8 waves/block (2/SIMD),
// branch-free B-operand construction via an LDS zero region, class loop
// unrolled x2 for LDS-load / WMMA overlap.
// ---------------------------------------------------------------------------

#define USE_ASYNC_LDS 1

typedef __attribute__((ext_vector_type(16))) __bf16 v16bf;
typedef __attribute__((ext_vector_type(8)))  float  v8f;

static __device__ __forceinline__ unsigned short f2bf_bits(float f) {
  unsigned u = __builtin_bit_cast(unsigned, f);
  u += 0x7fffu + ((u >> 16) & 1u);           // round-to-nearest-even
  return (unsigned short)(u >> 16);
}
static __device__ __forceinline__ __bf16 f2bf(float f) {
  return __builtin_bit_cast(__bf16, f2bf_bits(f));
}

static __device__ __forceinline__ v8f wmma_bf16(v16bf a, v16bf b, v8f c) {
  // emits v_wmma_f32_16x16x32_bf16
  return __builtin_amdgcn_wmma_f32_16x16x32_bf16(false, a, false, b, (short)0, c,
                                                 false, false);
}

static __device__ __forceinline__ v16bf load_b_operand(const char* p_, int es) {
  v16bf v = {};
#pragma unroll
  for (int s = 0; s < 8; s++) v[s] = *(const __bf16*)(p_ + s * es);
  return v;
}

#if USE_ASYNC_LDS
static __device__ __forceinline__ void async_copy_b128(void* lds, const void* gptr) {
  unsigned ldsaddr = (unsigned)(uintptr_t)lds;
  unsigned long long ga = (unsigned long long)(uintptr_t)gptr;
  asm volatile("global_load_async_to_lds_b128 %0, %1, off"
               :: "v"(ldsaddr), "v"(ga) : "memory");
}
static __device__ __forceinline__ void wait_async_zero() {
  asm volatile("s_wait_asynccnt 0" ::: "memory");
}
#endif

// ---------------- conv1 3x3 VALID + bias + relu : [2,1,64,64]->[2,128,62,62]
__global__ void k_conv1(const float* __restrict__ x, const float* __restrict__ w1,
                        const float* __restrict__ b1, float* __restrict__ h) {
  int c = threadIdx.x;                       // 0..127 output channel
  int oy = blockIdx.x, ox = blockIdx.y, b = blockIdx.z;
  const float* xs = x + b * 4096 + oy * 64 + ox;
  const float* wc = w1 + c * 9;
  float acc = b1[c];
#pragma unroll
  for (int ky = 0; ky < 3; ky++)
#pragma unroll
    for (int kx = 0; kx < 3; kx++)
      acc = fmaf(wc[ky * 3 + kx], xs[ky * 64 + kx], acc);
  h[((size_t)(b * 128 + c)) * 3844 + oy * 62 + ox] = fmaxf(acc, 0.f);
}

// ---------------- primary caps conv 3x3 s2 + bias : ->[2,128,30,30]
__global__ void k_pconv(const float* __restrict__ h, const float* __restrict__ pw,
                        const float* __restrict__ pb, float* __restrict__ p) {
  __shared__ float tile[128 * 9];            // input patch, all 128 channels
  int c = threadIdx.x;
  int oy = blockIdx.x, ox = blockIdx.y, b = blockIdx.z;
  const float* hs = h + ((size_t)(b * 128 + c)) * 3844 + (2 * oy) * 62 + 2 * ox;
#pragma unroll
  for (int ky = 0; ky < 3; ky++)
#pragma unroll
    for (int kx = 0; kx < 3; kx++)
      tile[c * 9 + ky * 3 + kx] = hs[ky * 62 + kx];
  __syncthreads();
  float acc = pb[c];
  const float* wc = pw + (size_t)c * 1152;
  for (int ci = 0; ci < 128; ci++) {
#pragma unroll
    for (int k = 0; k < 9; k++)
      acc = fmaf(tile[ci * 9 + k], wc[ci * 9 + k], acc);
  }
  p[((size_t)(b * 128 + c)) * 900 + oy * 30 + ox] = acc;
}

// ---------------- squash primary capsules: p[2,128,900] -> u[2,14400,8]
__global__ void k_squash_u(const float* __restrict__ p, float* __restrict__ u) {
  int idx = blockIdx.x * 256 + threadIdx.x;
  if (idx >= 2 * 14400) return;
  int b = idx / 14400, n = idx % 14400;
  int cap = n / 900, hw = n % 900;
  const float* ps = p + ((size_t)(b * 128 + cap * 8)) * 900 + hw;
  float v[8]; float sq = 0.f;
#pragma unroll
  for (int d = 0; d < 8; d++) { v[d] = ps[(size_t)d * 900]; sq = fmaf(v[d], v[d], sq); }
  float scale = sq / (1.f + sq) * rsqrtf(sq + 1e-8f);
  float* us = u + (size_t)idx * 8;
#pragma unroll
  for (int d = 0; d < 8; d++) us[d] = scale * v[d];
}

// ---------------- ubar[b,r,d] = sum_s u[b, 16*s+r, d]  (deterministic, 1 block)
__global__ void k_ubar(const float* __restrict__ u, float* __restrict__ ubar) {
  int t = threadIdx.x;                       // 256 = b(1).r(4).d(3)
  int b = t >> 7, r = (t >> 3) & 15, d = t & 7;
  float acc = 0.f;
  const float* us = u + ((size_t)b * 14400 + r) * 8 + d;
  for (int s = 0; s < 900; s++) acc += us[(size_t)s * 128];
  ubar[t] = acc;
}

// ---------------- iteration 0: uniform softmax => v0 from ubar (exact)
__global__ void k_iter0(const float* __restrict__ ubar, const float* __restrict__ W,
                        float* __restrict__ v0) {
  int c = blockIdx.x, b = blockIdx.y, e = threadIdx.x;   // 16 threads
  float s = 0.f;
  for (int r = 0; r < 16; r++)
#pragma unroll
    for (int d = 0; d < 8; d++)
      s = fmaf(ubar[(b * 16 + r) * 8 + d],
               W[(((size_t)(r * 276 + c)) * 8 + d) * 16 + e], s);
  s *= (1.f / 276.f);
  float sq = s * s;
#pragma unroll
  for (int m = 8; m >= 1; m >>= 1) sq += __shfl_xor(sq, m, 32);
  float scale = sq / (1.f + sq) * rsqrtf(sq + 1e-8f);
  v0[((size_t)(b * 276 + c)) * 16 + e] = scale * s;
}

// ---------------- W fp32 -> bf16 (same [r][c][d][e] layout)
__global__ void k_prepW(const float* __restrict__ W, unsigned short* __restrict__ wbf) {
  int idx = blockIdx.x * 256 + threadIdx.x;
  if (idx < 565248) wbf[idx] = f2bf_bits(W[idx]);
}

// ---------------- wv[b][r][d][c(288)] = sum_e W[r,c,d,e]*vsrc[b,c,e]  (bf16)
__global__ void k_wv(const float* __restrict__ W, const float* __restrict__ vsrc,
                     unsigned short* __restrict__ wvbf) {
  int idx = blockIdx.x * 256 + threadIdx.x;
  if (idx >= 2 * 16 * 8 * 288) return;
  int c = idx % 288;
  int d = (idx / 288) & 7;
  int r = (idx / 2304) & 15;
  int b = idx / 36864;
  float acc = 0.f;
  if (c < 276) {
    const float* wp = W + (((size_t)(r * 276 + c)) * 8 + d) * 16;
    const float* vp = vsrc + ((size_t)(b * 276 + c)) * 16;
#pragma unroll
    for (int e = 0; e < 16; e++) acc = fmaf(wp[e], vp[e], acc);
  }
  wvbf[idx] = f2bf_bits(acc);
}

// ---------------- routing pass: recompute u_hat via WMMA, softmax, accumulate s
// grid (8, r=16, b=2), block 256 (8 waves, 2 per SIMD). Each wave owns 16 n's
// sharing r. dyn LDS (240448 B total):
//   s_acc[276*16]f32 | logits[8][16][288]f32 | Wbf[276*8*16]bf16 | wv[8*288]bf16 | zeros
__global__ void k_route(const float* __restrict__ u, const unsigned short* __restrict__ wbf,
                        const unsigned short* __restrict__ wvbf,
                        float* __restrict__ partials) {
  extern __shared__ char smem[];
  float* s_acc  = (float*)smem;                                // [0, 17664)
  float* logits = (float*)(smem + 17664);                      // [17664, 165120)
  unsigned short* ldsW  = (unsigned short*)(smem + 165120);    // [165120, 235776)
  unsigned short* ldsWV = (unsigned short*)(smem + 235776);    // [235776, 240384)
  char* zerop = smem + 240384;                                 // 64B zero scratch

  const int tid = threadIdx.x, lane = tid & 31, w = tid >> 5;
  const int r = blockIdx.y, b = blockIdx.z;

  for (int i = tid; i < 4416; i += 256) s_acc[i] = 0.f;
  if (tid < 16) ((float*)zerop)[tid] = 0.f;

  // stage Wbf[r] (70.7 KB) + wv (4.6 KB) into LDS
  const unsigned short* gw = wbf + (size_t)r * 35328;
#if USE_ASYNC_LDS
  for (int i = tid; i < 4416; i += 256) async_copy_b128(ldsW + i * 8, gw + i * 8);
#else
  for (int i = tid; i < 4416; i += 256)
    *(uint4*)(ldsW + i * 8) = *(const uint4*)(gw + i * 8);
#endif
  const unsigned short* gwv = wvbf + (size_t)(b * 16 + r) * 2304;
  for (int i = tid; i < 288; i += 256)
    *(uint4*)(ldsWV + i * 8) = *(const uint4*)(gwv + i * 8);
#if USE_ASYNC_LDS
  wait_async_zero();
#endif
  __syncthreads();

  // A operand: 16 rows n = 16*(s0+m)+r, K=8 real (zero padded to 32)
  const int s0 = (blockIdx.x * 8 + w) * 16;
  const bool active = (s0 < 900);
  v16bf A = {};
  if (lane < 16 && (s0 + lane) < 900) {
    int n = 16 * (s0 + lane) + r;
    const float* up = u + ((size_t)b * 14400 + n) * 8;
#pragma unroll
    for (int k = 0; k < 8; k++) A[k] = f2bf(up[k]);
  }

  float* mylog = logits + w * (16 * 288);
  const int hi = lane >> 4;

  // branch-free B-operand addressing: lanes >= 16 always read the zero region
  const char* bbaseWV = (lane < 16) ? (const char*)(ldsWV + lane) : (const char*)zerop;
  const int   strdWV  = (lane < 16) ? 32 : 0;
  const char* bbaseW  = (lane < 16) ? (const char*)(ldsW + lane) : (const char*)zerop;
  const int   strdW   = (lane < 16) ? 256 : 0;

  // ---- pass 1: logits[n, c] = u . wv  (18 WMMAs covering 288 padded classes)
  if (active) {
#pragma unroll 2
    for (int g = 0; g < 18; g++) {
      v16bf Bv = load_b_operand(bbaseWV + g * strdWV, (lane < 16) ? 576 : 0);
      v8f acc = {};
      acc = wmma_bf16(A, Bv, acc);
      int cg = g * 16 + (lane & 15);
#pragma unroll
      for (int j = 0; j < 8; j++) mylog[(j + 8 * hi) * 288 + cg] = acc[j];
    }
  }
  __syncthreads();

  // ---- softmax over 276 classes per row (lane pairs split classes)
  if (active) {
    int row = lane & 15;
    float* lp = mylog + row * 288 + hi * 138;
    float mx = -1e30f;
    for (int i = 0; i < 138; i++) mx = fmaxf(mx, lp[i]);
    mx = fmaxf(mx, __shfl_xor(mx, 16, 32));
    float z = 0.f;
    for (int i = 0; i < 138; i++) { float ev = __expf(lp[i] - mx); lp[i] = ev; z += ev; }
    z += __shfl_xor(z, 16, 32);
    float inv = 1.f / z;
    for (int i = 0; i < 138; i++) lp[i] *= inv;
  }
  __syncthreads();

  // ---- pass 2: recompute u_hat per class, weight by coef, accumulate s
  if (active) {
#pragma unroll 2
    for (int c = 0; c < 276; c++) {
      v16bf Bv = load_b_operand(bbaseW + c * strdW, (lane < 16) ? 32 : 0);
      v8f acc = {};
      acc = wmma_bf16(A, Bv, acc);           // u_hat tile [16n x 16e] for class c
      float sp = 0.f;
#pragma unroll
      for (int j = 0; j < 8; j++)
        sp = fmaf(acc[j], mylog[(j + 8 * hi) * 288 + c], sp);
      sp += __shfl_xor(sp, 16, 32);          // sum rows across lane halves
      if (lane < 16) atomicAdd(&s_acc[c * 16 + lane], sp);   // ds_add_f32
    }
  }
  __syncthreads();

  const int blk = (blockIdx.z * 16 + blockIdx.y) * 8 + blockIdx.x;
  float* dst = partials + (size_t)blk * 4416;
  for (int i = tid; i < 4416; i += 256) dst[i] = s_acc[i];
}

// ---------------- reduce partials -> s, v = squash(s); optionally vsum / ||v||
__global__ void k_finish(const float* __restrict__ partials, float* __restrict__ vout,
                         const float* __restrict__ v0in, float* __restrict__ vsumout,
                         float* __restrict__ outp) {
  int bc = blockIdx.x;                       // 0..551
  int b = bc / 276, c = bc % 276;
  int tid = threadIdx.x;                     // 128
  int e = tid & 15, k = tid >> 4;
  float acc = 0.f;
  const float* base = partials + ((size_t)b * 128) * 4416 + c * 16 + e;
  for (int slot = k; slot < 128; slot += 8) acc += base[(size_t)slot * 4416];
  __shared__ float red[128];
  red[tid] = acc;
  __syncthreads();
  for (int st = 4; st >= 1; st >>= 1) {
    if (k < st) red[tid] += red[tid + st * 16];
    __syncthreads();
  }
  if (tid < 16) {
    float s = red[tid];
    float sq = s * s;
#pragma unroll
    for (int m = 8; m >= 1; m >>= 1) sq += __shfl_xor(sq, m, 32);
    float scale = sq / (1.f + sq) * rsqrtf(sq + 1e-8f);
    float v = scale * s;
    vout[(size_t)bc * 16 + tid] = v;
    if (vsumout) vsumout[(size_t)bc * 16 + tid] = v0in[(size_t)bc * 16 + tid] + v;
    if (outp && tid == 0) outp[bc] = scale * sqrtf(sq);   // ||v||
  }
}

// ---------------------------------------------------------------------------
extern "C" void kernel_launch(void* const* d_in, const int* in_sizes, int n_in,
                              void* d_out, int out_size, void* d_ws, size_t ws_size,
                              hipStream_t stream) {
  const float* x  = (const float*)d_in[0];
  const float* w1 = (const float*)d_in[1];
  const float* b1 = (const float*)d_in[2];
  const float* pw = (const float*)d_in[3];
  const float* pb = (const float*)d_in[4];
  const float* W  = (const float*)d_in[5];
  float* out = (float*)d_out;
  char* ws = (char*)d_ws;

  // workspace layout (bytes), total ~11.7 MB
  float* h    = (float*)(ws + 0);            // 2*128*62*62
  float* p    = (float*)(ws + 3936256);      // 2*128*900
  float* u    = (float*)(ws + 4857856);      // 2*14400*8
  float* ubar = (float*)(ws + 5779456);      // 2*16*8
  float* v0   = (float*)(ws + 5780480);      // 2*276*16
  float* v1   = (float*)(ws + 5815808);
  float* vsum = (float*)(ws + 5851136);
  unsigned short* wbf  = (unsigned short*)(ws + 5886464);   // 565248 bf16
  unsigned short* wvbf = (unsigned short*)(ws + 7016960);   // 73728 bf16
  float* part = (float*)(ws + 7164416);      // 256*4416 f32

  (void)hipFuncSetAttribute((const void*)k_route,
                            hipFuncAttributeMaxDynamicSharedMemorySize, 240448);

  k_conv1<<<dim3(62, 62, 2), 128, 0, stream>>>(x, w1, b1, h);
  k_pconv<<<dim3(30, 30, 2), 128, 0, stream>>>(h, pw, pb, p);
  k_squash_u<<<dim3(113), 256, 0, stream>>>(p, u);
  k_ubar<<<dim3(1), 256, 0, stream>>>(u, ubar);
  k_iter0<<<dim3(276, 2), 16, 0, stream>>>(ubar, W, v0);
  k_prepW<<<dim3(2208), 256, 0, stream>>>(W, wbf);

  // routing pass 1 (logits vs v0) -> v1, vsum = v0+v1
  k_wv<<<dim3(288), 256, 0, stream>>>(W, v0, wvbf);
  k_route<<<dim3(8, 16, 2), 256, 240448, stream>>>(u, wbf, wvbf, part);
  k_finish<<<dim3(552), 128, 0, stream>>>(part, v1, v0, vsum, nullptr);

  // routing pass 2 (logits vs v0+v1, exact by linearity) -> v2, output norms
  k_wv<<<dim3(288), 256, 0, stream>>>(W, vsum, wvbf);
  k_route<<<dim3(8, 16, 2), 256, 240448, stream>>>(u, wbf, wvbf, part);
  k_finish<<<dim3(552), 128, 0, stream>>>(part, v0, nullptr, nullptr, out);
}
